// GATBin_61125974557483
// MI455X (gfx1250) — matrix operand order repaired and checked by reading, not compiled, not run on hardware
//
#include <hip/hip_runtime.h>
#include <hip/hip_bf16.h>

// ---------------------------------------------------------------------------
// GATv2 x3 on MI455X (gfx1250).
//  - Dense node transforms: v_wmma_f32_16x16x32_f16, one wave per 16x16 tile.
//  - Edge passes: wave32-coalesced gather/scatter. Layer1 (HC=64): one wave
//    per edge, 2 channels/lane. Layer2 (HC=16): two edges per wave, 16
//    lanes/edge. Cross-lane head reductions via __shfl_xor butterflies.
//  - Softmax normalization folded into the node epilogue (divide by
//    segment-sum) to save one full edge pass.
// ---------------------------------------------------------------------------

typedef _Float16 v16h __attribute__((ext_vector_type(16)));
typedef _Float16 v8h  __attribute__((ext_vector_type(8)));
typedef float    v8f  __attribute__((ext_vector_type(8)));

#define NNODES 100000

// ---- float atomic max via sign-aware int/uint atomics ---------------------
__device__ __forceinline__ void atomicMaxFloat(float* addr, float val) {
  if (val >= 0.0f) atomicMax((int*)addr, __float_as_int(val));
  else             atomicMin((unsigned int*)addr, __float_as_uint(val));
}

// ---- generic fill ---------------------------------------------------------
__global__ void fill_f32_k(float* __restrict__ p, float v, long n) {
  long i = (long)blockIdx.x * blockDim.x + threadIdx.x;
  if (i < n) p[i] = v;
}

// ---- pack x (f32, row-major [N,K]) into padded f16 [N,Kp] -----------------
__global__ void pack_feat_k(const float* __restrict__ X, _Float16* __restrict__ Xh,
                            int N, int K, int Kp) {
  long idx = (long)blockIdx.x * blockDim.x + threadIdx.x;
  long total = (long)N * Kp;
  if (idx >= total) return;
  int n = (int)(idx / Kp), k = (int)(idx % Kp);
  Xh[idx] = (_Float16)((k < K) ? X[(long)n * K + k] : 0.0f);
}

// ---- pack W (f32 [K,Nout]) into transposed padded f16 [Npad,Kp] -----------
__global__ void pack_w_k(const float* __restrict__ W, _Float16* __restrict__ Wt,
                         int K, int Nout, int Kp, int Npad) {
  int idx = blockIdx.x * blockDim.x + threadIdx.x;
  int total = Npad * Kp;
  if (idx >= total) return;
  int n = idx / Kp, k = idx % Kp;
  float v = (k < K && n < Nout) ? W[(long)k * Nout + n] : 0.0f;
  Wt[idx] = (_Float16)v;
}

// ---- WMMA GEMM: out[M,Nout] = A[M,Kp](f16) * Wt[Npad,Kp]^T + bias ---------
// One wave per 16x16 tile. A/B/C lane layouts used here (A and B share contiguous k order):
//   A (16x32 f16): lanes 0-15 hold K 0-15; lanes 16-31 hold K 16-31
//   B (32x16 f16): lane column N=lane%16; lanes 0-15 K 0-15, lanes 16-31 K 16-31
//   C (16x16 f32): VGPR i -> row i (lanes 0-15) or i+8 (lanes 16-31)
__global__ void gemm_wmma_k(const _Float16* __restrict__ A,
                            const _Float16* __restrict__ Wt,
                            const float* __restrict__ bias,
                            float* __restrict__ out,
                            int M, int Kp, int Nout, int ntiles) {
  int wave = blockIdx.x * (blockDim.x >> 5) + (threadIdx.x >> 5);
  int lane = threadIdx.x & 31;
  int mtiles = M >> 4;
  if (wave >= mtiles * ntiles) return;
  int mt = wave / ntiles, nt = wave % ntiles;

  int mrow  = mt * 16 + (lane & 15);
  int ncol  = nt * 16 + (lane & 15);
  const _Float16* arow = A  + (long)mrow * Kp;
  const _Float16* brow = Wt + (long)ncol * Kp;
  const int aoff = (lane < 16) ? 0 : 16;  // halves
  const int boff = (lane < 16) ? 0 : 16;  // halves

  v8f c = {};
  for (int k0 = 0; k0 < Kp; k0 += 32) {
    union { v16h v; v8h h[2]; } ua;
    ua.h[0] = *(const v8h*)(arow + k0 + aoff);
    ua.h[1] = *(const v8h*)(arow + k0 + aoff + 8);
    v16h b = *(const v16h*)(brow + k0 + boff);
    c = __builtin_amdgcn_wmma_f32_16x16x32_f16(false, ua.v, false, b,
                                               (short)0, c, false, false);
  }

  if (ncol < Nout) {
    float bcol = bias[ncol];
    int mbase = mt * 16 + ((lane < 16) ? 0 : 8);
#pragma unroll
    for (int i = 0; i < 8; ++i)
      out[(long)(mbase + i) * Nout + ncol] = c[i] + bcol;
  }
}

// ===========================================================================
// Wave-coalesced edge pass 1: logits + running max per (dst, head).
// CPT = channels per thread; LPE = HC/CPT lanes per edge; EPW = 32/LPE edges
// per wave. Head group = C/CPT consecutive, power-of-two-aligned lanes, so
// __shfl_xor butterflies stay inside the group (and inside each sub-wave).
// ===========================================================================
template <int H, int C, int CPT>
__global__ void gat_edge_logits_w(const int* __restrict__ src, const int* __restrict__ dst,
                                  const float* __restrict__ xl, const float* __restrict__ xr,
                                  const float* __restrict__ att,
                                  float* __restrict__ logits, float* __restrict__ mx,
                                  int E) {
  const int HC  = H * C;
  const int LPE = HC / CPT;
  const int EPW = 32 / LPE;
  const int G   = C / CPT;            // lanes per head
  long wave = (long)blockIdx.x * (blockDim.x >> 5) + (threadIdx.x >> 5);
  int lane = threadIdx.x & 31;
  int l    = lane % LPE;              // lane within edge
  long e   = wave * EPW + lane / LPE;
  if (e >= E) return;
  int s = src[e], d = dst[e];
  int k0 = l * CPT;
  const float* pl = xl + (size_t)s * HC + k0;
  const float* pr = xr + (size_t)d * HC + k0;
  float partial = 0.0f;
#pragma unroll
  for (int j = 0; j < CPT; ++j) {
    float v = pl[j] + pr[j];
    v = (v > 0.0f) ? v : 0.2f * v;    // LeakyReLU(0.2)
    partial += att[k0 + j] * v;
  }
#pragma unroll
  for (int msk = G >> 1; msk > 0; msk >>= 1)
    partial += __shfl_xor(partial, msk, 32);
  if ((l & (G - 1)) == 0) {
    int h = k0 / C;
    logits[e * H + h] = partial;
    atomicMaxFloat(&mx[(size_t)d * H + h], partial);
  }
}

// ===========================================================================
// Wave-coalesced edge pass 2: a = exp(logit - m); denom += a (one lane per
// head); outacc[dst] += a * xl[src] with lane-consecutive addresses so the
// atomics coalesce into per-cacheline L2 bursts.
// ===========================================================================
template <int H, int C, int CPT>
__global__ void gat_edge_aggregate_w(const int* __restrict__ src, const int* __restrict__ dst,
                                     const float* __restrict__ xl,
                                     const float* __restrict__ logits,
                                     const float* __restrict__ mx,
                                     float* __restrict__ denom, float* __restrict__ outacc,
                                     int E) {
  const int HC  = H * C;
  const int LPE = HC / CPT;
  const int EPW = 32 / LPE;
  const int G   = C / CPT;
  long wave = (long)blockIdx.x * (blockDim.x >> 5) + (threadIdx.x >> 5);
  int lane = threadIdx.x & 31;
  int l    = lane % LPE;
  long e   = wave * EPW + lane / LPE;
  if (e >= E) return;
  int s = src[e], d = dst[e];
  int k0 = l * CPT;
  int h  = k0 / C;
  float a = expf(logits[e * H + h] - mx[(size_t)d * H + h]);
  if ((l & (G - 1)) == 0) atomicAdd(&denom[(size_t)d * H + h], a);
  const float* pl = xl + (size_t)s * HC + k0;
  float* po = outacc + (size_t)d * HC + k0;
#pragma unroll
  for (int j = 0; j < CPT; ++j)
    atomicAdd(&po[j], a * pl[j]);
}

// ---- scalar edge kernels for HC==1 (layer 3) ------------------------------
__global__ void gat_edge_logits_s(const int* __restrict__ src, const int* __restrict__ dst,
                                  const float* __restrict__ xl, const float* __restrict__ xr,
                                  const float* __restrict__ att,
                                  float* __restrict__ logits, float* __restrict__ mx, int E) {
  int e = blockIdx.x * blockDim.x + threadIdx.x;
  if (e >= E) return;
  int s = src[e], d = dst[e];
  float v = xl[s] + xr[d];
  v = (v > 0.0f) ? v : 0.2f * v;
  float lg = att[0] * v;
  logits[e] = lg;
  atomicMaxFloat(&mx[d], lg);
}

__global__ void gat_edge_aggregate_s(const int* __restrict__ src, const int* __restrict__ dst,
                                     const float* __restrict__ xl,
                                     const float* __restrict__ logits,
                                     const float* __restrict__ mx,
                                     float* __restrict__ denom, float* __restrict__ outacc,
                                     int E) {
  int e = blockIdx.x * blockDim.x + threadIdx.x;
  if (e >= E) return;
  int s = src[e], d = dst[e];
  float a = expf(logits[e] - mx[d]);
  atomicAdd(&denom[d], a);
  atomicAdd(&outacc[d], a * xl[s]);
}

// ---- node epilogue (layers 1/2): normalize, +bias, ReLU, emit padded f16 --
template <int H, int C>
__global__ void gat_node_epilogue(const float* __restrict__ outacc,
                                  const float* __restrict__ denom,
                                  const float* __restrict__ bias,
                                  _Float16* __restrict__ featH, int N, int Kp) {
  int n = blockIdx.x * blockDim.x + threadIdx.x;
  if (n >= N) return;
  const int HC = H * C;
#pragma unroll
  for (int k = 0; k < HC; ++k) {
    int h = k / C;
    float v = outacc[(size_t)n * HC + k] / (denom[(size_t)n * H + h] + 1e-16f) + bias[k];
    featH[(size_t)n * Kp + k] = (_Float16)fmaxf(v, 0.0f);
  }
  for (int k = HC; k < Kp; ++k) featH[(size_t)n * Kp + k] = (_Float16)0.0f;
}

// ---- final epilogue: H=1,C=1, mean==identity; softmax over 1 channel == 1 -
__global__ void gat_final_k(const float* __restrict__ outacc,
                            const float* __restrict__ denom,
                            const float* __restrict__ bias,
                            float* __restrict__ out, int N) {
  int n = blockIdx.x * blockDim.x + threadIdx.x;
  if (n >= N) return;
  float v = outacc[n] / (denom[n] + 1e-16f) + bias[0];
  v = fmaxf(v, 0.0f);
  out[n] = expf(v - v);   // softmax over a single channel is identically 1
}

// ---------------------------------------------------------------------------
extern "C" void kernel_launch(void* const* d_in, const int* in_sizes, int n_in,
                              void* d_out, int out_size, void* d_ws, size_t ws_size,
                              hipStream_t stream) {
  const int N = NNODES;
  const float* x   = (const float*)d_in[0];
  const int*   ei  = (const int*)d_in[1];
  const int    E   = in_sizes[1] / 2;
  const int* src = ei;
  const int* dst = ei + E;
  const float *Wl1 = (const float*)d_in[2],  *bl1 = (const float*)d_in[3];
  const float *Wr1 = (const float*)d_in[4],  *br1 = (const float*)d_in[5];
  const float *att1 = (const float*)d_in[6], *bias1 = (const float*)d_in[7];
  const float *Wl2 = (const float*)d_in[8],  *bl2 = (const float*)d_in[9];
  const float *Wr2 = (const float*)d_in[10], *br2 = (const float*)d_in[11];
  const float *att2 = (const float*)d_in[12], *bias2 = (const float*)d_in[13];
  const float *Wlf = (const float*)d_in[14], *blf = (const float*)d_in[15];
  const float *Wrf = (const float*)d_in[16], *brf = (const float*)d_in[17];
  const float *attf = (const float*)d_in[18], *biasf = (const float*)d_in[19];
  float* outp = (float*)d_out;

  // ---- workspace carve-out (256B aligned) ----
  char* ws = (char*)d_ws;
  size_t off = 0;
  auto carve = [&](size_t bytes) -> void* {
    void* p = ws + off;
    off += (bytes + 255) & ~(size_t)255;
    return p;
  };
  _Float16* featH  = (_Float16*)carve((size_t)N * 128 * 2); // f16 features (reused per layer)
  float*    xl     = (float*)carve((size_t)N * 64 * 4);
  float*    xr     = (float*)carve((size_t)N * 64 * 4);
  float*    outacc = (float*)carve((size_t)N * 64 * 4);
  float*    m      = (float*)carve((size_t)N * 8 * 4);
  float*    denom  = (float*)carve((size_t)N * 8 * 4);
  float*    logits = (float*)carve((size_t)E * 8 * 4);
  _Float16* Wl1t = (_Float16*)carve(64 * 128 * 2);
  _Float16* Wr1t = (_Float16*)carve(64 * 128 * 2);
  _Float16* Wl2t = (_Float16*)carve(16 * 64 * 2);
  _Float16* Wr2t = (_Float16*)carve(16 * 64 * 2);
  _Float16* Wlft = (_Float16*)carve(16 * 32 * 2);
  _Float16* Wrft = (_Float16*)carve(16 * 32 * 2);
  (void)ws_size;

  const int TB = 256;
  auto blocks = [](long n, int tb) { return (int)((n + tb - 1) / tb); };
  // blocks for wave-granular kernels: 8 waves per 256-thread block
  auto wblocks = [](long waves) { return (int)((waves + 7) / 8); };
  const float NEGINF = -__builtin_huge_valf();

  // ---- pack x and all weights to f16 ----
  pack_feat_k<<<blocks((long)N * 128, TB), TB, 0, stream>>>(x, featH, N, 128, 128);
  pack_w_k<<<blocks(64 * 128, TB), TB, 0, stream>>>(Wl1, Wl1t, 128, 64, 128, 64);
  pack_w_k<<<blocks(64 * 128, TB), TB, 0, stream>>>(Wr1, Wr1t, 128, 64, 128, 64);
  pack_w_k<<<blocks(16 * 64, TB), TB, 0, stream>>>(Wl2, Wl2t, 64, 16, 64, 16);
  pack_w_k<<<blocks(16 * 64, TB), TB, 0, stream>>>(Wr2, Wr2t, 64, 16, 64, 16);
  pack_w_k<<<blocks(16 * 32, TB), TB, 0, stream>>>(Wlf, Wlft, 16, 1, 32, 16);
  pack_w_k<<<blocks(16 * 32, TB), TB, 0, stream>>>(Wrf, Wrft, 16, 1, 32, 16);

  const int mtiles = N / 16;                 // 6250
  auto gemm_blocks = [&](int ntiles) { return (mtiles * ntiles + 7) / 8; }; // 8 waves/block

  // ================= layer 1: in=128, H=4, C=16, HC=64 =================
  // edge mapping: 1 edge/wave, 32 lanes, 2 channels/lane
  gemm_wmma_k<<<gemm_blocks(4), TB, 0, stream>>>(featH, Wl1t, bl1, xl, N, 128, 64, 4);
  gemm_wmma_k<<<gemm_blocks(4), TB, 0, stream>>>(featH, Wr1t, br1, xr, N, 128, 64, 4);
  fill_f32_k<<<blocks((long)N * 4, TB), TB, 0, stream>>>(m, NEGINF, (long)N * 4);
  fill_f32_k<<<blocks((long)N * 4, TB), TB, 0, stream>>>(denom, 0.0f, (long)N * 4);
  fill_f32_k<<<blocks((long)N * 64, TB), TB, 0, stream>>>(outacc, 0.0f, (long)N * 64);
  gat_edge_logits_w<4, 16, 2><<<wblocks(E), TB, 0, stream>>>(src, dst, xl, xr, att1, logits, m, E);
  gat_edge_aggregate_w<4, 16, 2><<<wblocks(E), TB, 0, stream>>>(src, dst, xl, logits, m, denom, outacc, E);
  gat_node_epilogue<4, 16><<<blocks(N, TB), TB, 0, stream>>>(outacc, denom, bias1, featH, N, 64);

  // ================= layer 2: in=64, H=8, C=2, HC=16 ==================
  // edge mapping: 2 edges/wave, 16 lanes/edge, 1 channel/lane
  gemm_wmma_k<<<gemm_blocks(1), TB, 0, stream>>>(featH, Wl2t, bl2, xl, N, 64, 16, 1);
  gemm_wmma_k<<<gemm_blocks(1), TB, 0, stream>>>(featH, Wr2t, br2, xr, N, 64, 16, 1);
  fill_f32_k<<<blocks((long)N * 8, TB), TB, 0, stream>>>(m, NEGINF, (long)N * 8);
  fill_f32_k<<<blocks((long)N * 8, TB), TB, 0, stream>>>(denom, 0.0f, (long)N * 8);
  fill_f32_k<<<blocks((long)N * 16, TB), TB, 0, stream>>>(outacc, 0.0f, (long)N * 16);
  gat_edge_logits_w<8, 2, 1><<<wblocks((E + 1) / 2), TB, 0, stream>>>(src, dst, xl, xr, att2, logits, m, E);
  gat_edge_aggregate_w<8, 2, 1><<<wblocks((E + 1) / 2), TB, 0, stream>>>(src, dst, xl, logits, m, denom, outacc, E);
  gat_node_epilogue<8, 2><<<blocks(N, TB), TB, 0, stream>>>(outacc, denom, bias2, featH, N, 32);

  // ================= layer 3: in=16 (padded 32), H=1, C=1 =============
  gemm_wmma_k<<<gemm_blocks(1), TB, 0, stream>>>(featH, Wlft, blf, xl, N, 32, 1, 1);
  gemm_wmma_k<<<gemm_blocks(1), TB, 0, stream>>>(featH, Wrft, brf, xr, N, 32, 1, 1);
  fill_f32_k<<<blocks(N, TB), TB, 0, stream>>>(m, NEGINF, N);
  fill_f32_k<<<blocks(N, TB), TB, 0, stream>>>(denom, 0.0f, N);
  fill_f32_k<<<blocks(N, TB), TB, 0, stream>>>(outacc, 0.0f, N);
  gat_edge_logits_s<<<blocks(E, TB), TB, 0, stream>>>(src, dst, xl, xr, attf, logits, m, E);
  gat_edge_aggregate_s<<<blocks(E, TB), TB, 0, stream>>>(src, dst, xl, logits, m, denom, outacc, E);
  gat_final_k<<<blocks(N, TB), TB, 0, stream>>>(outacc, denom, biasf, outp, N);
}
